// BVAE_38783554683545
// MI455X (gfx1250) — compile-verified, matrix operand order checked by reading, not run on hardware
//
#include <hip/hip_runtime.h>
#include <hip/hip_bf16.h>

typedef __attribute__((ext_vector_type(16))) _Float16 v16h;
typedef __attribute__((ext_vector_type(8)))  _Float16 v8h;
typedef __attribute__((ext_vector_type(8)))  float    v8f;

#define ROWS   16384   // T_MC * BATCH
#define D_IN   784
#define Z_DIM  40
#define NBASIS 7
#define NMCMC  10000

constexpr int BM = 128, BN = 64, BK = 32, LDK = BK + 8;  // 40-half rows: 16B aligned, bank-spread

// ---------------------------------------------------------------------------
// Double-buffered WMMA GEMM core.
//   TA: A element type (float or _Float16)  -- f16 path copies straight to LDS
//   TO: C element type (float or _Float16)  -- f16 output fuses the convert
//   EDGE=false: block interior in M/N; K-tail via address-clamp + select only.
// ---------------------------------------------------------------------------
template<int ACT, bool EDGE, typename TA, typename TO>
__device__ __forceinline__ void gemm_core(
    const TA* __restrict__ A, const float* __restrict__ W,
    const float* __restrict__ bias, TO* __restrict__ C,
    int M, int N, int K, int m0, int n0,
    _Float16 (&As)[2][BM][LDK], _Float16 (&Bs)[2][BN][LDK])
{
  constexpr bool A16 = (sizeof(TA) == 2);

  const int tid  = threadIdx.x;
  const int lane = tid & 31;
  const int wave = tid >> 5;
  const int wm = wave & 3, wn = wave >> 2;   // 4x2 wave grid
  const int grp = lane >> 4, l16 = lane & 15;

  // staging assignments: A -> 16 contiguous elems/thread, B -> 2x4 floats
  const int arow = tid >> 1, acol = (tid & 1) * 16;
  const int bk   = tid >> 4, bq   = (tid & 15) * 4;

  const int  gmA = EDGE ? ((m0 + arow < M) ? (m0 + arow) : (M - 1)) : (m0 + arow);
  const bool mOK = EDGE ? (m0 + arow < M) : true;

  const int ktFull = K / BK;
  const int kTiles = (K + BK - 1) / BK;

  float4 ra[4];        // f32-A staging regs
  v8h    ha[2];        // f16-A staging regs
  float4 rb[2];

  // ---- global -> registers for tile kt ----
  auto gload = [&](int kt) {
    const int k0 = kt * BK;
    if (!EDGE && kt < ktFull) {            // guard-free vectorized path
      if constexpr (A16) {
        const v8h* srcA = reinterpret_cast<const v8h*>(A + (size_t)gmA * K + k0 + acol);
        ha[0] = srcA[0]; ha[1] = srcA[1];
      } else {
        const float4* srcA = reinterpret_cast<const float4*>(A + (size_t)gmA * K + k0 + acol);
        ra[0] = srcA[0]; ra[1] = srcA[1]; ra[2] = srcA[2]; ra[3] = srcA[3];
      }
      rb[0] = *reinterpret_cast<const float4*>(W + (size_t)(k0 + bk) * N + n0 + bq);
      rb[1] = *reinterpret_cast<const float4*>(W + (size_t)(k0 + bk + 16) * N + n0 + bq);
      if (kt + 2 < ktFull) {               // speculative prefetch 2 tiles ahead
        __builtin_prefetch(A + (size_t)gmA * K + k0 + 2 * BK + acol, 0, 1);
        __builtin_prefetch(W + (size_t)(k0 + 2 * BK + bk) * N + n0 + bq, 0, 1);
      }
    } else {                               // clamped + selected (no exec masks)
      if constexpr (A16) {
        #pragma unroll
        for (int c = 0; c < 16; ++c) {
          const int gk  = k0 + acol + c;
          const int gkc = (gk < K) ? gk : 0;
          const _Float16 v = A[(size_t)gmA * K + gkc];
          ha[c >> 3][c & 7] = (mOK && gk < K) ? v : (_Float16)0.0f;
        }
      } else {
        float ta[16];
        #pragma unroll
        for (int c = 0; c < 16; ++c) {
          const int gk  = k0 + acol + c;
          const int gkc = (gk < K) ? gk : 0;
          const float v = A[(size_t)gmA * K + gkc];
          ta[c] = (mOK && gk < K) ? v : 0.0f;
        }
        ra[0] = make_float4(ta[0],  ta[1],  ta[2],  ta[3]);
        ra[1] = make_float4(ta[4],  ta[5],  ta[6],  ta[7]);
        ra[2] = make_float4(ta[8],  ta[9],  ta[10], ta[11]);
        ra[3] = make_float4(ta[12], ta[13], ta[14], ta[15]);
      }
      float tb[8];
      #pragma unroll
      for (int rr = 0; rr < 2; ++rr) {
        const int gk  = k0 + bk + rr * 16;
        const int gkc = (gk < K) ? gk : 0;
        #pragma unroll
        for (int c = 0; c < 4; ++c) {
          const int gn  = n0 + bq + c;
          const int gnc = EDGE ? ((gn < N) ? gn : 0) : gn;
          const float v = W[(size_t)gkc * N + gnc];
          tb[rr * 4 + c] = (gk < K && (!EDGE || gn < N)) ? v : 0.0f;
        }
      }
      rb[0] = make_float4(tb[0], tb[1], tb[2], tb[3]);
      rb[1] = make_float4(tb[4], tb[5], tb[6], tb[7]);
    }
  };

  // ---- registers -> LDS buffer ----
  auto sstore = [&](int buf) {
    if constexpr (A16) {
      *reinterpret_cast<v8h*>(&As[buf][arow][acol])     = ha[0];
      *reinterpret_cast<v8h*>(&As[buf][arow][acol + 8]) = ha[1];
    } else {
      v8h h0, h1;
      h0[0] = (_Float16)ra[0].x; h0[1] = (_Float16)ra[0].y;
      h0[2] = (_Float16)ra[0].z; h0[3] = (_Float16)ra[0].w;
      h0[4] = (_Float16)ra[1].x; h0[5] = (_Float16)ra[1].y;
      h0[6] = (_Float16)ra[1].z; h0[7] = (_Float16)ra[1].w;
      h1[0] = (_Float16)ra[2].x; h1[1] = (_Float16)ra[2].y;
      h1[2] = (_Float16)ra[2].z; h1[3] = (_Float16)ra[2].w;
      h1[4] = (_Float16)ra[3].x; h1[5] = (_Float16)ra[3].y;
      h1[6] = (_Float16)ra[3].z; h1[7] = (_Float16)ra[3].w;
      *reinterpret_cast<v8h*>(&As[buf][arow][acol])     = h0;
      *reinterpret_cast<v8h*>(&As[buf][arow][acol + 8]) = h1;
    }
    Bs[buf][bq + 0][bk]      = (_Float16)rb[0].x;
    Bs[buf][bq + 1][bk]      = (_Float16)rb[0].y;
    Bs[buf][bq + 2][bk]      = (_Float16)rb[0].z;
    Bs[buf][bq + 3][bk]      = (_Float16)rb[0].w;
    Bs[buf][bq + 0][bk + 16] = (_Float16)rb[1].x;
    Bs[buf][bq + 1][bk + 16] = (_Float16)rb[1].y;
    Bs[buf][bq + 2][bk + 16] = (_Float16)rb[1].z;
    Bs[buf][bq + 3][bk + 16] = (_Float16)rb[1].w;
  };

  v8f acc[2][2] = {};

  // ---- fragment loads + 4 WMMAs on LDS buffer ----
  auto compute = [&](int buf) {
    v16h afrag[2], bfrag[2];
    #pragma unroll
    for (int f = 0; f < 2; ++f) {
      const int ar = wm * 32 + f * 16 + l16;
      const v8h alo = *reinterpret_cast<const v8h*>(&As[buf][ar][grp * 8]);
      const v8h ahi = *reinterpret_cast<const v8h*>(&As[buf][ar][16 + grp * 8]);
      const int bc = wn * 32 + f * 16 + l16;
      const v8h blo = *reinterpret_cast<const v8h*>(&Bs[buf][bc][grp * 8]);
      const v8h bhi = *reinterpret_cast<const v8h*>(&Bs[buf][bc][16 + grp * 8]);
      #pragma unroll
      for (int h = 0; h < 8; ++h) {
        afrag[f][h] = alo[h];  afrag[f][8 + h] = ahi[h];
        bfrag[f][h] = blo[h];  bfrag[f][8 + h] = bhi[h];
      }
    }
    #pragma unroll
    for (int fm = 0; fm < 2; ++fm)
      #pragma unroll
      for (int fn = 0; fn < 2; ++fn)
        acc[fm][fn] = __builtin_amdgcn_wmma_f32_16x16x32_f16(
            false, afrag[fm], false, bfrag[fn],
            (short)0, acc[fm][fn], false, false);
  };

  // ---- software pipeline: load(kt+1) | compute(kt) | store(kt+1) | barrier
  gload(0);
  sstore(0);
  __syncthreads();
  int cur = 0;
  for (int kt = 0; kt < kTiles; ++kt) {
    const bool more = (kt + 1 < kTiles);
    if (more) gload(kt + 1);       // global loads issue early
    compute(cur);                  // ds_loads + wmma overlap load latency
    if (more) sstore(cur ^ 1);     // loadcnt waited here, not at compute
    __syncthreads();
    cur ^= 1;
  }

  // ---- epilogue: C/D layout VGPR r -> lanes 0-15 M=r, lanes 16-31 M=8+r ----
  #pragma unroll
  for (int fm = 0; fm < 2; ++fm) {
    #pragma unroll
    for (int fn = 0; fn < 2; ++fn) {
      const int colBase = n0 + wn * 32 + fn * 16 + l16;
      const float bv = EDGE ? ((colBase < N) ? bias[colBase] : 0.0f) : bias[colBase];
      #pragma unroll
      for (int r = 0; r < 8; ++r) {
        const int row = m0 + wm * 32 + fm * 16 + grp * 8 + r;
        float v = acc[fm][fn][r] + bv;
        if (ACT == 1)       v = (v > 0.0f) ? v : (__expf(v) - 1.0f);
        else if (ACT == 2)  v = 1.0f / (1.0f + __expf(-v));
        if (!EDGE || (row < M && colBase < N))
          C[(size_t)row * N + colBase] = (TO)v;
      }
    }
  }
}

template<int ACT, typename TA, typename TO>
__global__ __launch_bounds__(256) void gemm_bias_act(
    const TA* __restrict__ A, const float* __restrict__ W,
    const float* __restrict__ bias, TO* __restrict__ C,
    int M, int N, int K)
{
  __shared__ alignas(16) _Float16 As[2][BM][LDK];
  __shared__ alignas(16) _Float16 Bs[2][BN][LDK];
  const int m0 = blockIdx.y * BM;
  const int n0 = blockIdx.x * BN;
  if (m0 + BM <= M && n0 + BN <= N)
    gemm_core<ACT, false>(A, W, bias, C, M, N, K, m0, n0, As, Bs);
  else
    gemm_core<ACT, true >(A, W, bias, C, M, N, K, m0, n0, As, Bs);
}

// ---------------------------------------------------------------------------
// Degree-3 B-spline basis (Cox-de Boor) at point x for the fixed knot vector.
// ---------------------------------------------------------------------------
__device__ __forceinline__ void bspline7(float x, float* out) {
  const float t[11] = {0.f,0.f,0.f,0.f,0.25f,0.5f,0.75f,1.f,1.f,1.f,1.f};
  float N[10];
  #pragma unroll
  for (int i = 0; i < 10; ++i)
    N[i] = (x >= t[i] && x < t[i + 1]) ? 1.0f : 0.0f;
  #pragma unroll
  for (int d = 1; d <= 3; ++d) {
    #pragma unroll
    for (int i = 0; i < 9; ++i) {
      if (i < 10 - d) {
        const float den_l = t[i + d]     - t[i];
        const float den_r = t[i + d + 1] - t[i + 1];
        const float left  = (den_l > 0.f) ? (x - t[i]) / den_l * N[i] : 0.f;
        const float right = (den_r > 0.f) ? (t[i + d + 1] - x) / den_r * N[i + 1] : 0.f;
        N[i] = left + right;   // forward in-place: N[i+1] still degree d-1
      }
    }
  }
  #pragma unroll
  for (int i = 0; i < NBASIS; ++i) out[i] = N[i];
}

// ---------------------------------------------------------------------------
// Per-latent sampling: one thread per (t,b,z) element (ROWS*Z_DIM = 655360).
// ---------------------------------------------------------------------------
__global__ __launch_bounds__(256) void sampling_kernel(
    const float* __restrict__ lat, const float* __restrict__ gumbel,
    const int* __restrict__ mcmc_idx, const float* __restrict__ basis_mcmc,
    float* __restrict__ coef_out, float* __restrict__ weights_out,
    float* __restrict__ z_out, float* __restrict__ pdf_out,
    float* __restrict__ zstd_out)
{
  const int idx = blockIdx.x * blockDim.x + threadIdx.x;
  if (idx >= ROWS * Z_DIM) return;

  const float Iinv[NBASIS] = {16.f, 8.f, 16.f/3.f, 4.f, 16.f/3.f, 8.f, 16.f};

  const float* l = lat + (size_t)idx * (NBASIS + 2);
  const float mu = l[0], logvar = l[1];
  const float zstd = __expf(0.5f * logvar);

  // softmax over 7 spline logits
  float lg[NBASIS], mx = -3.4e38f;
  #pragma unroll
  for (int i = 0; i < NBASIS; ++i) { lg[i] = l[2 + i]; mx = fmaxf(mx, lg[i]); }
  float e[NBASIS], s = 0.f;
  #pragma unroll
  for (int i = 0; i < NBASIS; ++i) { e[i] = __expf(lg[i] - mx); s += e[i]; }
  const float invs = 1.0f / s, logZ = __logf(s);
  float coef[NBASIS], cw[NBASIS];
  #pragma unroll
  for (int i = 0; i < NBASIS; ++i) { coef[i] = e[i] * invs; cw[i] = coef[i] * Iinv[i]; }

  // Gumbel-softmax indicator (1/TEMP = 10), log(coef) = lg - mx - logZ
  const float* g = gumbel + (size_t)idx * NBASIS;
  float il[NBASIS], mx2 = -3.4e38f;
  #pragma unroll
  for (int i = 0; i < NBASIS; ++i) {
    il[i] = ((lg[i] - mx - logZ) + g[i]) * 10.0f;
    mx2 = fmaxf(mx2, il[i]);
  }
  float ind[NBASIS], s2 = 0.f;
  #pragma unroll
  for (int i = 0; i < NBASIS; ++i) { ind[i] = __expf(il[i] - mx2); s2 += ind[i]; }
  const float inv2 = 1.0f / s2;
  #pragma unroll
  for (int i = 0; i < NBASIS; ++i) ind[i] *= inv2;

  // MCMC table gather + sample mixing
  const int* mi = mcmc_idx + (size_t)idx * NBASIS;
  float bs[NBASIS], spl = 0.f;
  #pragma unroll
  for (int i = 0; i < NBASIS; ++i) {
    bs[i] = basis_mcmc[i * NMCMC + mi[i]];
    spl += ind[i] * bs[i];
  }

  // pdf_approx = sum_j ind[j] * sum_i (coef[i]/I[i]) * N_i(bs[j])
  float pdf = 0.f;
  #pragma unroll
  for (int j = 0; j < NBASIS; ++j) {
    float n7[NBASIS];
    bspline7(bs[j], n7);
    float sj = 0.f;
    #pragma unroll
    for (int i = 0; i < NBASIS; ++i) sj += cw[i] * n7[i];
    pdf += ind[j] * sj;
  }

  // writes
  #pragma unroll
  for (int i = 0; i < NBASIS; ++i) {
    coef_out   [(size_t)idx * NBASIS + i] = coef[i];
    weights_out[(size_t)idx * NBASIS + i] = cw[i];
  }
  z_out   [idx] = spl * zstd + mu;
  pdf_out [idx] = pdf;
  zstd_out[idx] = zstd;
}

// ---------------------------------------------------------------------------
extern "C" void kernel_launch(void* const* d_in, const int* in_sizes, int n_in,
                              void* d_out, int out_size, void* d_ws, size_t ws_size,
                              hipStream_t stream)
{
  (void)in_sizes; (void)n_in; (void)out_size; (void)ws_size;

  const float* x          = (const float*)d_in[0];
  const float* We0        = (const float*)d_in[1];
  const float* be0        = (const float*)d_in[2];
  const float* We1        = (const float*)d_in[3];
  const float* be1        = (const float*)d_in[4];
  const float* We2        = (const float*)d_in[5];
  const float* be2        = (const float*)d_in[6];
  const float* Wd0        = (const float*)d_in[7];
  const float* bd0        = (const float*)d_in[8];
  const float* Wd1        = (const float*)d_in[9];
  const float* bd1        = (const float*)d_in[10];
  const float* Wd2        = (const float*)d_in[11];
  const float* bd2        = (const float*)d_in[12];
  const float* Wv0        = (const float*)d_in[13];
  const float* bv0        = (const float*)d_in[14];
  const float* Wv1        = (const float*)d_in[15];
  const float* bv1        = (const float*)d_in[16];
  const float* Wv2        = (const float*)d_in[17];
  const float* bv2        = (const float*)d_in[18];
  const float* basis_mcmc = (const float*)d_in[19];
  const float* gumbel     = (const float*)d_in[20];
  const int*   mcmc_idx   = (const int*)  d_in[21];

  // scratch (48 MB total, comfortably L2-resident):
  //   hbufA: ROWS x 1024 f16 (32 MB)  -- h1 / hm2 / hv2 ; lat (f32) overlays it
  //   hbufB: ROWS x  512 f16 (16 MB)  -- h2 / hm1 / hv1
  _Float16* hbufA = (_Float16*)d_ws;
  _Float16* hbufB = hbufA + (size_t)ROWS * 1024;
  float*    lat   = (float*)d_ws;      // ROWS x 360 f32 (22.5 MB) overlays hbufA

  // output layout (reference tuple, flat concat)
  float* out          = (float*)d_out;
  float* recon_mean   = out;
  float* recon_var    = recon_mean  + (size_t)ROWS * D_IN;
  float* coef_out     = recon_var   + (size_t)ROWS * D_IN;
  float* weights_out  = coef_out    + (size_t)ROWS * Z_DIM * NBASIS;
  float* z_out        = weights_out + (size_t)ROWS * Z_DIM * NBASIS;
  float* pdf_out      = z_out       + (size_t)ROWS * Z_DIM;
  float* zstd_out     = pdf_out     + (size_t)ROWS * Z_DIM;

  const dim3 blk(256);
  auto grid = [](int M, int N) { return dim3((N + 63) / 64, (M + 127) / 128); };

  // encoder: x(f32) -> h1(f16) -> h2(f16) -> lat(f32)
  gemm_bias_act<1, float,    _Float16><<<grid(ROWS, 1024), blk, 0, stream>>>(
      x,     We0, be0, hbufA, ROWS, 1024, D_IN);
  gemm_bias_act<1, _Float16, _Float16><<<grid(ROWS,  512), blk, 0, stream>>>(
      hbufA, We1, be1, hbufB, ROWS,  512, 1024);
  gemm_bias_act<0, _Float16, float   ><<<grid(ROWS, (NBASIS + 2) * Z_DIM), blk, 0, stream>>>(
      hbufB, We2, be2, lat,   ROWS, (NBASIS + 2) * Z_DIM, 512);

  // sampling / spline machinery (reads f32 lat, writes 5 outputs)
  sampling_kernel<<<(ROWS * Z_DIM + 255) / 256, 256, 0, stream>>>(
      lat, gumbel, mcmc_idx, basis_mcmc,
      coef_out, weights_out, z_out, pdf_out, zstd_out);

  // decoder mean tower: z(f32,K=40) -> hm1(f16) -> hm2(f16) -> recon_mean(f32)
  gemm_bias_act<1, float,    _Float16><<<grid(ROWS,  512), blk, 0, stream>>>(
      z_out, Wd0, bd0, hbufB, ROWS,  512, Z_DIM);
  gemm_bias_act<1, _Float16, _Float16><<<grid(ROWS, 1024), blk, 0, stream>>>(
      hbufB, Wd1, bd1, hbufA, ROWS, 1024, 512);
  gemm_bias_act<2, _Float16, float   ><<<grid(ROWS, D_IN), blk, 0, stream>>>(
      hbufA, Wd2, bd2, recon_mean, ROWS, D_IN, 1024);

  // decoder variance tower
  gemm_bias_act<1, float,    _Float16><<<grid(ROWS,  512), blk, 0, stream>>>(
      z_out, Wv0, bv0, hbufB, ROWS,  512, Z_DIM);
  gemm_bias_act<1, _Float16, _Float16><<<grid(ROWS, 1024), blk, 0, stream>>>(
      hbufB, Wv1, bv1, hbufA, ROWS, 1024, 512);
  gemm_bias_act<2, _Float16, float   ><<<grid(ROWS, D_IN), blk, 0, stream>>>(
      hbufA, Wv2, bv2, recon_var, ROWS, D_IN, 1024);
}
